// GNNModel_32220844654999
// MI455X (gfx1250) — matrix operand order repaired and checked by reading, not compiled, not run on hardware
//
#include <hip/hip_runtime.h>
#include <math.h>

// ---------------------------------------------------------------------------
// CDNA5 (gfx1250) WMMA types
// ---------------------------------------------------------------------------
typedef __attribute__((ext_vector_type(16))) __bf16        v16bf;
typedef __attribute__((ext_vector_type(8)))  float         v8f;
typedef __attribute__((ext_vector_type(8)))  unsigned int  v8u;

#define EPI_NONE        0
#define EPI_BIAS        1
#define EPI_BIAS_LRELU  2
#define EPI_RELU_ACC    3   // C[idx] += acc; relu
#define EPI_BIAS_SELU   4

#define LRELU_SLOPE 0.01f
#define SELU_ALPHA  1.6732632423543772848170429916717f
#define SELU_SCALE  1.0507009873554804934193349852946f

__device__ __forceinline__ float selu_f(float x) {
    return SELU_SCALE * (x > 0.f ? x : SELU_ALPHA * (expf(x) - 1.f));
}

__device__ __forceinline__ unsigned pack_bf16(float lo, float hi) {
    union { __bf16 b[2]; unsigned u; } t;
    t.b[0] = (__bf16)lo;
    t.b[1] = (__bf16)hi;
    return t.u;   // -> v_cvt_pk_bf16_f32
}

// ---------------------------------------------------------------------------
// bf16-WMMA GEMM:  C[M,N] = epi( A[M,K] @ B[K,N] (+bias / +C) )
//   - A,B fp32 row-major; converted to bf16 while staged into LDS (float4 loads)
//   - REQUIRES K % 32 == 0 (true for all GEMMs in this model: 64/128/256)
//   - Block: 256 thr = 8 wave32; block tile 64(M) x 128(N); K-step 32
//   - Each wave: 32x32 patch = 2x2 WMMA tiles -> 4 x v_wmma_f32_16x16x32_bf16
//     per K-step, A/B fragments each reused twice.
//   - B is staged transposed (Bt[n][k]) so both fragment types are contiguous
//     dword runs in LDS.
// ---------------------------------------------------------------------------
#define BM 64
#define BN 128
#define BK 32
#define ALD 17   // dword stride per A row  (34 bf16: 32 + 2 pad, 4B-aligned rows)
#define BLD 17   // dword stride per Bt row

__global__ __launch_bounds__(256)
void wmma_gemm_kernel(const float* __restrict__ A, const float* __restrict__ B,
                      float* __restrict__ C, const float* __restrict__ bias,
                      int M, int N, int K, int epi) {
    __shared__ unsigned As32[BM * ALD];   // [m][k-pair]
    __shared__ unsigned Bt32[BN * BLD];   // [n][k-pair]  (B transposed)

    const int tid  = threadIdx.x;
    const int lane = tid & 31;
    const int wave = tid >> 5;
    const int wm   = wave & 1;      // 0..1 : 32-row slab
    const int wn   = wave >> 1;     // 0..3 : 32-col slab
    const int blockM = blockIdx.y * BM;
    const int blockN = blockIdx.x * BN;

    const int mrow  = lane & 15;    // row (A) / col (B) within 16-tile
    const int khalf = lane >> 4;    // K-half group per CDNA5 16-bit layouts

    v8f acc[2][2] = {};

    // A staging map: one 8-float run per thread: row = tid/4, col = (tid%4)*8
    const int arow = tid >> 2;
    const int acol = (tid & 3) * 8;
    const bool arow_ok = (blockM + arow) < M;
    const float* Aptr = A + (size_t)(blockM + arow) * K + acol;

    for (int k0 = 0; k0 < K; k0 += BK) {
        // ---- stage A (64 x 32) ----
        {
            float4 v0 = make_float4(0.f, 0.f, 0.f, 0.f), v1 = v0;
            if (arow_ok) {
                v0 = *(const float4*)(Aptr + k0);
                v1 = *(const float4*)(Aptr + k0 + 4);
            }
            unsigned* d = &As32[arow * ALD + (acol >> 1)];
            d[0] = pack_bf16(v0.x, v0.y);
            d[1] = pack_bf16(v0.z, v0.w);
            d[2] = pack_bf16(v1.x, v1.y);
            d[3] = pack_bf16(v1.z, v1.w);
        }
        // ---- stage B (32 x 128) transposed ----
        {
            __bf16* Bt16 = (__bf16*)Bt32;
            for (int t = tid; t < (BK * BN / 4); t += 256) {
                int kk = t >> 5;          // 0..31
                int n  = (t & 31) * 4;    // 0..124
                int gn = blockN + n;
                const float* bp = &B[(size_t)(k0 + kk) * N + gn];
                float4 v = make_float4(0.f, 0.f, 0.f, 0.f);
                if (gn + 3 < N) {
                    v = *(const float4*)bp;
                } else {
                    if (gn + 0 < N) v.x = bp[0];
                    if (gn + 1 < N) v.y = bp[1];
                    if (gn + 2 < N) v.z = bp[2];
                }
                Bt16[(n + 0) * (2 * BLD) + kk] = (__bf16)v.x;
                Bt16[(n + 1) * (2 * BLD) + kk] = (__bf16)v.y;
                Bt16[(n + 2) * (2 * BLD) + kk] = (__bf16)v.z;
                Bt16[(n + 3) * (2 * BLD) + kk] = (__bf16)v.w;
            }
        }
        __syncthreads();

        // ---- fragments (contiguous dword runs) ----
        // A 16x32: lanes 0-15 K0..7 & 16..23, lanes 16-31 K8..15 & 24..31
        v16bf afrag[2], bfrag[2];
#pragma unroll
        for (int sm = 0; sm < 2; ++sm) {
            const unsigned* Ar = &As32[(wm * 32 + sm * 16 + mrow) * ALD];
            v8u u;
#pragma unroll
            for (int j = 0; j < 4; ++j) {
                u[j]     = Ar[khalf * 4 + j];
                u[4 + j] = Ar[8 + khalf * 4 + j];
            }
            afrag[sm] = __builtin_bit_cast(v16bf, u);
        }
        // B 32x16: lane = column; lanes 0-15 K0..7 & 16..23, lanes 16-31 K8..15 & 24..31
#pragma unroll
        for (int sn = 0; sn < 2; ++sn) {
            const unsigned* Br = &Bt32[(wn * 32 + sn * 16 + mrow) * BLD];
            v8u u;
#pragma unroll
            for (int j = 0; j < 4; ++j) {
                u[j]     = Br[khalf * 4 + j];
                u[4 + j] = Br[8 + khalf * 4 + j];
            }
            bfrag[sn] = __builtin_bit_cast(v16bf, u);
        }

        // ---- 4 WMMAs, no divergence (EXEC all-ones) ----
#pragma unroll
        for (int sm = 0; sm < 2; ++sm)
#pragma unroll
            for (int sn = 0; sn < 2; ++sn)
                acc[sm][sn] = __builtin_amdgcn_wmma_f32_16x16x32_bf16(
                    false, afrag[sm], false, bfrag[sn], (short)0,
                    acc[sm][sn], false, false);
        __syncthreads();
    }

    // ---- epilogue / store.  C/D: VGPR r -> M = r + 8*khalf, N = lane%16 ----
#pragma unroll
    for (int sm = 0; sm < 2; ++sm) {
        const int mbase = blockM + wm * 32 + sm * 16 + khalf * 8;
#pragma unroll
        for (int sn = 0; sn < 2; ++sn) {
            const int ncol = blockN + wn * 32 + sn * 16 + mrow;
            if (ncol >= N) continue;
            const bool has_bias =
                (epi == EPI_BIAS) | (epi == EPI_BIAS_LRELU) | (epi == EPI_BIAS_SELU);
            const float bv = has_bias ? bias[ncol] : 0.f;
#pragma unroll
            for (int r = 0; r < 8; ++r) {
                const int m = mbase + r;
                if (m < M) {
                    size_t idx = (size_t)m * N + ncol;
                    float v = acc[sm][sn][r];
                    if (epi == EPI_RELU_ACC)        { v += C[idx]; v = fmaxf(v, 0.f); }
                    else if (epi == EPI_BIAS)       { v += bv; }
                    else if (epi == EPI_BIAS_LRELU) { v += bv; v = v > 0.f ? v : LRELU_SLOPE * v; }
                    else if (epi == EPI_BIAS_SELU)  { v += bv; v = selu_f(v); }
                    C[idx] = v;
                }
            }
        }
    }
}

// ---------------------------------------------------------------------------
// Graph aggregation kernels (memory-bound; plain VALU + atomics)
// ---------------------------------------------------------------------------
__global__ void degree_kernel(const int* __restrict__ idx, int E, float* __restrict__ deg) {
    int e = blockIdx.x * blockDim.x + threadIdx.x;
    if (e < E) atomicAdd(&deg[idx[e]], 1.0f);
}

__global__ void scatter_add_kernel(const float* __restrict__ hsrc,
                                   const int* __restrict__ src, const int* __restrict__ dst,
                                   int E, int d, float* __restrict__ out) {
    const int chunks = d >> 2;   // d is a multiple of 4
    long long i = (long long)blockIdx.x * blockDim.x + threadIdx.x;
    long long total = (long long)E * chunks;
    if (i >= total) return;
    int e = (int)(i / chunks);
    int c = (int)(i % chunks) * 4;
    const float4 v = *(const float4*)&hsrc[(size_t)src[e] * d + c];
    float* o = &out[(size_t)dst[e] * d + c];
    atomicAdd(o + 0, v.x); atomicAdd(o + 1, v.y);
    atomicAdd(o + 2, v.z); atomicAdd(o + 3, v.w);
}

__global__ void divide_deg_kernel(float* __restrict__ x, const float* __restrict__ deg,
                                  int n, int d) {
    long long i = (long long)blockIdx.x * blockDim.x + threadIdx.x;
    if (i >= (long long)n * d) return;
    int row = (int)(i / d);
    x[i] = x[i] / fmaxf(deg[row], 1.0f);
}

__global__ __launch_bounds__(256)
void rownorm_kernel(float* __restrict__ z, int n, int d) {
    __shared__ float red[256];
    int row = blockIdx.x;
    float s = 0.f;
    for (int c = threadIdx.x; c < d; c += 256) {
        float v = z[(size_t)row * d + c];
        s += v * v;
    }
    red[threadIdx.x] = s;
    __syncthreads();
    for (int off = 128; off > 0; off >>= 1) {
        if (threadIdx.x < off) red[threadIdx.x] += red[threadIdx.x + off];
        __syncthreads();
    }
    float nrm = sqrtf(red[0]);
    float div = (nrm == 0.f) ? 1.f : nrm;
    for (int c = threadIdx.x; c < d; c += 256)
        z[(size_t)row * d + c] /= div;
}

__global__ void add_inplace_kernel(float* __restrict__ out, const float* __restrict__ a,
                                   long long n) {
    long long i = (long long)blockIdx.x * blockDim.x + threadIdx.x;
    if (i < n) out[i] += a[i];
}

__global__ void gather_concat_kernel(const float* __restrict__ hc, const float* __restrict__ hp,
                                     const int* __restrict__ s, const int* __restrict__ dd,
                                     int E, int d, float* __restrict__ x) {
    long long i = (long long)blockIdx.x * blockDim.x + threadIdx.x;
    long long total = (long long)E * (2 * d);
    if (i >= total) return;
    int e = (int)(i / (2 * d));
    int c = (int)(i % (2 * d));
    x[i] = (c < d) ? hc[(size_t)s[e] * d + c] : hp[(size_t)dd[e] * d + (c - d)];
}

// y1[E,16] already SELU'd; apply 16->8 (SELU) -> 1
__global__ void pred_tail_kernel(const float* __restrict__ y1,
                                 const float* __restrict__ W2, const float* __restrict__ b2,
                                 const float* __restrict__ W3, const float* __restrict__ b3,
                                 int E, float* __restrict__ out) {
    int e = blockIdx.x * blockDim.x + threadIdx.x;
    if (e >= E) return;
    float x1[16];
#pragma unroll
    for (int i = 0; i < 16; ++i) x1[i] = y1[(size_t)e * 16 + i];
    float acc3 = b3[0];
#pragma unroll
    for (int j = 0; j < 8; ++j) {
        float a = b2[j];
#pragma unroll
        for (int i = 0; i < 16; ++i) a += x1[i] * W2[i * 8 + j];
        acc3 += selu_f(a) * W3[j];
    }
    out[e] = acc3;
}

// ---------------------------------------------------------------------------
// Host orchestration
// ---------------------------------------------------------------------------
static inline int cdivi(long long a, long long b) { return (int)((a + b - 1) / b); }

extern "C" void kernel_launch(void* const* d_in, const int* in_sizes, int n_in,
                              void* d_out, int out_size, void* d_ws, size_t ws_size,
                              hipStream_t stream) {
    (void)n_in; (void)out_size; (void)ws_size;
    const int D_C = 128, D_P = 128, D_G = 64, EMBD = 128, HID = 256;
    const int N_C = in_sizes[0] / D_C;
    const int N_P = in_sizes[1] / D_P;
    const int N_G = in_sizes[2] / D_G;

    const float* cf = (const float*)d_in[0];
    const float* pf = (const float*)d_in[1];
    const float* gf = (const float*)d_in[2];

    // params flattened in JAX pytree (sorted-dict-key) order:
    // conv1{cp,gp,pc,pg}{Wn,Ws}, conv2{...}, cust{W1,W2,b1,b2}, grp{...},
    // pred{W1,W2,W3,b1,b2,b3}, prod{W1,W2,b1,b2}
    const float* c1_cp_Wn = (const float*)d_in[3];
    const float* c1_cp_Ws = (const float*)d_in[4];
    const float* c1_gp_Wn = (const float*)d_in[5];
    const float* c1_gp_Ws = (const float*)d_in[6];
    const float* c1_pc_Wn = (const float*)d_in[7];
    const float* c1_pc_Ws = (const float*)d_in[8];
    const float* c1_pg_Wn = (const float*)d_in[9];
    const float* c1_pg_Ws = (const float*)d_in[10];
    const float* c2_cp_Wn = (const float*)d_in[11];
    const float* c2_cp_Ws = (const float*)d_in[12];
    const float* c2_gp_Wn = (const float*)d_in[13];
    const float* c2_gp_Ws = (const float*)d_in[14];
    const float* c2_pc_Wn = (const float*)d_in[15];
    const float* c2_pc_Ws = (const float*)d_in[16];
    const float* c2_pg_Wn = (const float*)d_in[17];
    const float* c2_pg_Ws = (const float*)d_in[18];
    const float* cu_W1 = (const float*)d_in[19];
    const float* cu_W2 = (const float*)d_in[20];
    const float* cu_b1 = (const float*)d_in[21];
    const float* cu_b2 = (const float*)d_in[22];
    const float* gr_W1 = (const float*)d_in[23];
    const float* gr_W2 = (const float*)d_in[24];
    const float* gr_b1 = (const float*)d_in[25];
    const float* gr_b2 = (const float*)d_in[26];
    const float* pr_W1 = (const float*)d_in[27];
    const float* pr_W2 = (const float*)d_in[28];
    const float* pr_W3 = (const float*)d_in[29];
    const float* pr_b1 = (const float*)d_in[30];
    const float* pr_b2 = (const float*)d_in[31];
    const float* pr_b3 = (const float*)d_in[32];
    const float* po_W1 = (const float*)d_in[33];
    const float* po_W2 = (const float*)d_in[34];
    const float* po_b1 = (const float*)d_in[35];
    const float* po_b2 = (const float*)d_in[36];

    const int* cp_src = (const int*)d_in[37];
    const int* cp_dst = (const int*)d_in[38];
    const int* pg_src = (const int*)d_in[39];
    const int* pg_dst = (const int*)d_in[40];
    const int* pos_src = (const int*)d_in[41];
    const int* pos_dst = (const int*)d_in[42];
    const int* neg_src = (const int*)d_in[43];
    const int* neg_dst = (const int*)d_in[44];
    const int E_CP = in_sizes[37];
    const int E_PG = in_sizes[39];
    const int E_PR = in_sizes[41];

    // workspace layout (floats)
    float* ws = (float*)d_ws;
    size_t o = 0;
    float* hc    = ws + o; o += (size_t)N_C * HID;
    float* hp    = ws + o; o += (size_t)N_P * HID;
    float* hg    = ws + o; o += (size_t)N_G * HID;
    float* nc    = ws + o; o += (size_t)N_C * HID;
    float* np_   = ws + o; o += (size_t)N_P * HID;
    float* ng    = ws + o; o += (size_t)N_G * HID;
    float* neigh = ws + o; o += (size_t)N_P * HID;          // max n_dst*din
    float* tmp   = ws + o; o += (size_t)E_PR * (2 * EMBD);  // embed scratch / z2 / xcat
    float* y1    = ws + o; o += (size_t)E_PR * 16;
    float* deg_c  = ws + o; o += N_C;
    float* deg_pc = ws + o; o += N_P;
    float* deg_pg = ws + o; o += N_P;
    float* deg_g  = ws + o; o += N_G;

    float* out_hc  = (float*)d_out;
    float* out_hp  = out_hc + (size_t)N_C * EMBD;
    float* out_hg  = out_hp + (size_t)N_P * EMBD;
    float* out_pos = out_hg + (size_t)N_G * EMBD;
    float* out_neg = out_pos + E_PR;

    auto gemm = [&](const float* A, const float* B, float* C, const float* bias,
                    int M, int N, int K, int epi) {
        dim3 grid((unsigned)cdivi(N, BN), (unsigned)cdivi(M, BM));
        wmma_gemm_kernel<<<grid, 256, 0, stream>>>(A, B, C, bias, M, N, K, epi);
    };

    // z = rownorm(relu(hdst@Ws + mean_scatter(hsrc[src]->dst)@Wn))
    auto conv = [&](const float* Ws_, const float* Wn_,
                    const float* hsrc, const float* hdst,
                    const int* src, const int* dst, const float* deg,
                    int E, int n_dst, int din, int dout, float* z) {
        hipMemsetAsync(neigh, 0, (size_t)n_dst * din * sizeof(float), stream);
        long long tot = (long long)E * (din / 4);
        scatter_add_kernel<<<cdivi(tot, 256), 256, 0, stream>>>(hsrc, src, dst, E, din, neigh);
        divide_deg_kernel<<<cdivi((long long)n_dst * din, 256), 256, 0, stream>>>(neigh, deg, n_dst, din);
        gemm(hdst, Ws_, z, nullptr, n_dst, dout, din, EPI_NONE);
        gemm(neigh, Wn_, z, nullptr, n_dst, dout, din, EPI_RELU_ACC);
        rownorm_kernel<<<n_dst, 256, 0, stream>>>(z, n_dst, dout);
    };

    // ---- node embeddings: h = lrelu((x@W1+b1)@W2+b2) ----
    gemm(cf, cu_W1, tmp, cu_b1, N_C, EMBD, D_C, EPI_BIAS);
    gemm(tmp, cu_W2, hc, cu_b2, N_C, HID, EMBD, EPI_BIAS_LRELU);
    gemm(pf, po_W1, tmp, po_b1, N_P, EMBD, D_P, EPI_BIAS);
    gemm(tmp, po_W2, hp, po_b2, N_P, HID, EMBD, EPI_BIAS_LRELU);
    gemm(gf, gr_W1, tmp, gr_b1, N_G, EMBD, D_G, EPI_BIAS);
    gemm(tmp, gr_W2, hg, gr_b2, N_G, HID, EMBD, EPI_BIAS_LRELU);

    // ---- degrees (constant across layers) ----
    hipMemsetAsync(deg_c, 0, (size_t)(N_C + 2 * N_P + N_G) * sizeof(float), stream);
    degree_kernel<<<cdivi(E_CP, 256), 256, 0, stream>>>(cp_src, E_CP, deg_c);
    degree_kernel<<<cdivi(E_CP, 256), 256, 0, stream>>>(cp_dst, E_CP, deg_pc);
    degree_kernel<<<cdivi(E_PG, 256), 256, 0, stream>>>(pg_src, E_PG, deg_pg);
    degree_kernel<<<cdivi(E_PG, 256), 256, 0, stream>>>(pg_dst, E_PG, deg_g);

    // ---- conv layer 1 (256 -> 256) ----
    conv(c1_pc_Ws, c1_pc_Wn, hp, hc, cp_dst, cp_src, deg_c, E_CP, N_C, HID, HID, nc);
    conv(c1_cp_Ws, c1_cp_Wn, hc, hp, cp_src, cp_dst, deg_pc, E_CP, N_P, HID, HID, np_);
    conv(c1_gp_Ws, c1_gp_Wn, hg, hp, pg_dst, pg_src, deg_pg, E_PG, N_P, HID, HID, tmp);
    add_inplace_kernel<<<cdivi((long long)N_P * HID, 256), 256, 0, stream>>>(np_, tmp, (long long)N_P * HID);
    conv(c1_pg_Ws, c1_pg_Wn, hp, hg, pg_src, pg_dst, deg_g, E_PG, N_G, HID, HID, ng);

    // ---- conv layer 2 (256 -> 128), write into d_out ----
    conv(c2_pc_Ws, c2_pc_Wn, np_, nc, cp_dst, cp_src, deg_c, E_CP, N_C, HID, EMBD, out_hc);
    conv(c2_cp_Ws, c2_cp_Wn, nc, np_, cp_src, cp_dst, deg_pc, E_CP, N_P, HID, EMBD, out_hp);
    conv(c2_gp_Ws, c2_gp_Wn, ng, np_, pg_dst, pg_src, deg_pg, E_PG, N_P, HID, EMBD, tmp);
    add_inplace_kernel<<<cdivi((long long)N_P * EMBD, 256), 256, 0, stream>>>(out_hp, tmp, (long long)N_P * EMBD);
    conv(c2_pg_Ws, c2_pg_Wn, np_, ng, pg_src, pg_dst, deg_g, E_PG, N_G, HID, EMBD, out_hg);

    // ---- predictor ----
    long long gtot = (long long)E_PR * (2 * EMBD);
    // positive edges
    gather_concat_kernel<<<cdivi(gtot, 256), 256, 0, stream>>>(out_hc, out_hp, pos_src, pos_dst, E_PR, EMBD, tmp);
    gemm(tmp, pr_W1, y1, pr_b1, E_PR, 16, 2 * EMBD, EPI_BIAS_SELU);
    pred_tail_kernel<<<cdivi(E_PR, 256), 256, 0, stream>>>(y1, pr_W2, pr_b2, pr_W3, pr_b3, E_PR, out_pos);
    // negative edges
    gather_concat_kernel<<<cdivi(gtot, 256), 256, 0, stream>>>(out_hc, out_hp, neg_src, neg_dst, E_PR, EMBD, tmp);
    gemm(tmp, pr_W1, y1, pr_b1, E_PR, 16, 2 * EMBD, EPI_BIAS_SELU);
    pred_tail_kernel<<<cdivi(E_PR, 256), 256, 0, stream>>>(y1, pr_W2, pr_b2, pr_W3, pr_b3, E_PR, out_neg);
}
